// SpatialConv_58961311039782
// MI455X (gfx1250) — compile-verified, matrix operand order checked by reading, not run on hardware
//
#include <hip/hip_runtime.h>

// ---------------------------------------------------------------------------
// 5-tap cross conv (KERNEL_TYPE=0), N=16, Ci=Co=256, H=W=64, fp32 in/out.
// Implicit GEMM, M=Co, N=pixels(one image row), K=Ci*5, computed with
// v_wmma_f32_16x16x32_bf16 using the hi/lo bf16 split (3 products) for
// near-fp32 accuracy at bf16 matrix-pipe rate.
// ---------------------------------------------------------------------------

typedef __attribute__((ext_vector_type(16))) __bf16 v16bf;
typedef __attribute__((ext_vector_type(8)))  float  v8f;

#define NBATCH 16
#define CI     256
#define CO     256
#define HH     64
#define WW     64
#define NTAPS  5

// LDS activation tile: 2 planes (hi/lo) x 3 rows x 66 cols (w halo) x 32 ch
#define LDS_PLANE (3 * 66 * 32)      // halves per plane = 6336
#define LDS_TOTAL (2 * LDS_PLANE)    // 12672 halves = 25344 B

// Weight fragment workspace: 16 mtiles * 5 taps * 8 ctiles = 640 tiles,
// x 2 planes x 32 lanes x 8 dwords = 1,310,720 bytes (must fit in d_ws).
#define WS_TILES   (16 * NTAPS * 8)
#define WS_DWORDS  ((size_t)WS_TILES * 2 * 32 * 8)

__device__ __forceinline__ unsigned short f2bf(float f) {
  unsigned u = __builtin_bit_cast(unsigned, f);
  u += 0x7FFFu + ((u >> 16) & 1u);            // round-to-nearest-even
  return (unsigned short)(u >> 16);
}
__device__ __forceinline__ float bf2f(unsigned short b) {
  return __builtin_bit_cast(float, ((unsigned)b) << 16);
}

__device__ __forceinline__ v16bf ld_frag_lds(const unsigned short* p) {
  union { uint4 q[2]; v16bf v; } u;
  u.q[0] = *(const uint4*)(p);        // channels 0..7   (of this K half)
  u.q[1] = *(const uint4*)(p + 16);   // channels 16..23 (of this K half)
  return u.v;
}
__device__ __forceinline__ v16bf ld_frag_glb(const uint4* p) {
  union { uint4 q[2]; v16bf v; } u;
  u.q[0] = p[0];
  u.q[1] = p[1];
  return u.v;
}

// ---------------------------------------------------------------------------
// Pre-pack weights (Co,Ci,5) fp32 -> per-lane WMMA A-fragments, hi & lo bf16
// planes, laid out so the main kernel loads one fragment as 2x b128.
// One thread per (tile, plane, lane) produces 8 dwords.
// ---------------------------------------------------------------------------
__global__ __launch_bounds__(256) void prepack_weights(
    const float* __restrict__ w, unsigned int* __restrict__ wsA) {
  int gid   = blockIdx.x * 256 + threadIdx.x;     // 0 .. 40959
  int lane  = gid & 31;
  int plane = (gid >> 5) & 1;
  int tile  = gid >> 6;                           // 0 .. 639
  int ctile = tile & 7;
  int tap   = (tile >> 3) % NTAPS;
  int mtile = tile / (NTAPS * 8);
  int co    = mtile * 16 + (lane & 15);
  int kh    = lane >> 4;                          // K-half select

  unsigned int d[8];
#pragma unroll
  for (int v = 0; v < 8; ++v) {
    int cb = ctile * 32 + (v & 3) * 2 + 8 * kh + 16 * (v >> 2);
    float w0 = w[(size_t)co * (CI * NTAPS) + (size_t)cb * NTAPS + tap];
    float w1 = w[(size_t)co * (CI * NTAPS) + (size_t)(cb + 1) * NTAPS + tap];
    unsigned short h0 = f2bf(w0), h1 = f2bf(w1);
    if (plane == 0) {
      d[v] = ((unsigned)h1 << 16) | h0;
    } else {
      unsigned short l0 = f2bf(w0 - bf2f(h0));
      unsigned short l1 = f2bf(w1 - bf2f(h1));
      d[v] = ((unsigned)l1 << 16) | l0;
    }
  }
  uint4* q = (uint4*)(wsA + (size_t)gid * 8);
  q[0] = make_uint4(d[0], d[1], d[2], d[3]);
  q[1] = make_uint4(d[4], d[5], d[6], d[7]);
}

// ---------------------------------------------------------------------------
// Main conv kernel. Block = 256 threads = 8 waves (4 Co-waves x 2 px-waves).
// Block tile: 128 Co x 64 pixels (one (n,h) output row).
// Grid: 1024 rows * 2 Co halves = 2048 blocks.
// ---------------------------------------------------------------------------
__global__ __launch_bounds__(256) void conv_cross_wmma(
    const float* __restrict__ x, const unsigned int* __restrict__ wsA,
    float* __restrict__ out) {
  // taps: k0 center, k1 up, k2 left, k3 right, k4 down
  constexpr int tapR[NTAPS] = {1, 0, 1, 1, 2};   // staged-row index (h-1,h,h+1)
  constexpr int tapX[NTAPS] = {1, 1, 0, 2, 1};   // w halo column offset

  __shared__ __align__(16) unsigned short s_x[LDS_TOTAL];

  const int tid  = threadIdx.x;
  const int lane = tid & 31;
  const int wid  = tid >> 5;
  const int wm   = wid & 3;                       // Co wave (0..3)
  const int wn   = wid >> 2;                      // pixel wave (0..1)
  const int kh   = lane >> 4;                     // K-half / M-half select
  const int nl   = lane & 15;

  const int bx  = blockIdx.x;
  const int mb  = bx & 1;                         // Co half (0..1)
  const int row = bx >> 1;                        // 0..1023
  const int n   = row >> 6;
  const int oh  = row & 63;

  const int mtile0 = mb * 8 + wm * 2;             // first 16-Co tile of wave

  // zero LDS once (data region rewritten each chunk; halo stays zero)
  for (int i = tid; i < LDS_TOTAL / 2; i += 256)
    ((unsigned int*)s_x)[i] = 0u;
  __syncthreads();

  v8f acc[2][2] = {};

  for (int ct = 0; ct < 8; ++ct) {                // Ci chunks of 32
    const int cbase = ct * 32;

    // ---- stage x[n, cbase..cbase+31, oh-1..oh+1, :] as bf16 hi/lo ----
    for (int e = tid; e < 3 * 32 * 64; e += 256) {
      int r  = e >> 11;          // staged row 0..2
      int c  = (e >> 6) & 31;    // channel in chunk
      int wx = e & 63;           // column
      int hy = oh + r - 1;
      if (0 <= hy && hy < HH) {
        size_t gi = (((size_t)n * CI + cbase + c) * HH + hy) * WW + wx;
        float v = x[gi];
        if (ct < 7) __builtin_prefetch(&x[gi + (size_t)32 * HH * WW], 0, 1);
        unsigned short hb = f2bf(v);
        unsigned short lb = f2bf(v - bf2f(hb));
        int o = (r * 66 + (wx + 1)) * 32 + c;
        s_x[o]             = hb;
        s_x[LDS_PLANE + o] = lb;
      }
    }
    __syncthreads();

    // ---- compute: 5 taps x 2x2 tiles x 3 split-products of WMMA ----
#pragma unroll
    for (int tap = 0; tap < NTAPS; ++tap) {
      v16bf aH[2], aL[2];
#pragma unroll
      for (int ms = 0; ms < 2; ++ms) {
        int tile = ((mtile0 + ms) * NTAPS + tap) * 8 + ct;
        const uint4* ph = (const uint4*)wsA + ((size_t)(tile * 2 + 0) * 32 + lane) * 2;
        const uint4* pl = (const uint4*)wsA + ((size_t)(tile * 2 + 1) * 32 + lane) * 2;
        aH[ms] = ld_frag_glb(ph);
        aL[ms] = ld_frag_glb(pl);
      }
      v16bf bH[2], bL[2];
#pragma unroll
      for (int ns = 0; ns < 2; ++ns) {
        int wpix = wn * 32 + ns * 16 + nl;
        int base = (tapR[tap] * 66 + (wpix + tapX[tap])) * 32 + 8 * kh;
        bH[ns] = ld_frag_lds(&s_x[base]);
        bL[ns] = ld_frag_lds(&s_x[LDS_PLANE + base]);
      }
#pragma unroll
      for (int ms = 0; ms < 2; ++ms)
#pragma unroll
        for (int ns = 0; ns < 2; ++ns) {
          acc[ms][ns] = __builtin_amdgcn_wmma_f32_16x16x32_bf16(
              false, aH[ms], false, bH[ns], (short)0, acc[ms][ns], false, false);
          acc[ms][ns] = __builtin_amdgcn_wmma_f32_16x16x32_bf16(
              false, aH[ms], false, bL[ns], (short)0, acc[ms][ns], false, false);
          acc[ms][ns] = __builtin_amdgcn_wmma_f32_16x16x32_bf16(
              false, aL[ms], false, bH[ns], (short)0, acc[ms][ns], false, false);
        }
    }
    __syncthreads();
  }

  // ---- epilogue: C/D layout -> out[n, co, oh, w] ----
  const int cob = mb * 128 + wm * 32;
  const size_t outbase = ((size_t)n * CO) * (HH * WW) + (size_t)oh * WW;
#pragma unroll
  for (int ms = 0; ms < 2; ++ms)
#pragma unroll
    for (int ns = 0; ns < 2; ++ns) {
      int wpix = wn * 32 + ns * 16 + nl;
#pragma unroll
      for (int j = 0; j < 8; ++j) {
        int co = cob + ms * 16 + j + 8 * kh;
        out[outbase + (size_t)co * (HH * WW) + wpix] = acc[ms][ns][j];
      }
    }
}

// ---------------------------------------------------------------------------
extern "C" void kernel_launch(void* const* d_in, const int* in_sizes, int n_in,
                              void* d_out, int out_size, void* d_ws, size_t ws_size,
                              hipStream_t stream) {
  const float* x = (const float*)d_in[0];
  const float* w = (const float*)d_in[1];
  float* out = (float*)d_out;
  unsigned int* ws = (unsigned int*)d_ws;   // needs >= 1,310,720 bytes

  // 1) pack weights into WMMA A-fragment hi/lo bf16 layout
  prepack_weights<<<WS_TILES * 2 * 32 / 256, 256, 0, stream>>>(w, ws);
  // 2) main conv: 1024 output rows x 2 Co halves
  conv_cross_wmma<<<(NBATCH * HH) * 2, 256, 0, stream>>>(x, ws, out);
}
